// GumbelSort_68367289418254
// MI455X (gfx1250) — compile-verified, matrix operand order checked by reading, not run on hardware
//
#include <hip/hip_runtime.h>
#include <hip/hip_bf16.h>
#include <math.h>

#define N_TOTAL 16384
#define CHUNK   2048
#define NCHUNK  (N_TOTAL / CHUNK)      // 8
#define BLOCK   128                    // 4 wave32s
#define WAVES   (BLOCK / 32)
#define ROWS_PER_WAVE 16
#define GRID    (N_TOTAL / (WAVES * ROWS_PER_WAVE))  // 256 blocks

typedef __attribute__((ext_vector_type(16))) _Float16 v16h;
typedef __attribute__((ext_vector_type(8)))  _Float16 v8h;
typedef __attribute__((ext_vector_type(4)))  float    v4f;
typedef __attribute__((ext_vector_type(8)))  float    v8f;
typedef int vec_b128 __attribute__((vector_size(16)));  // async builtin's pointee

#if __has_builtin(__builtin_amdgcn_global_load_async_to_lds_b128) && \
    __has_builtin(__builtin_amdgcn_s_wait_asynccnt)
#define HAVE_ASYNC 1
#else
#define HAVE_ASYNC 0
#endif

// ---------------- kernel 1: g = l + noise ; e = exp(l) (f16) ----------------
__global__ void gumbel_prep(const float* __restrict__ logits,
                            const float* __restrict__ noise,
                            float* __restrict__ g_out,
                            _Float16* __restrict__ e_out) {
    int i = blockIdx.x * blockDim.x + threadIdx.x;
    float l = logits[i];
    g_out[i] = l + noise[i];
    e_out[i] = (_Float16)expf(l);
}

// ---------------- kernel 2: masked row-sums via WMMA ------------------------
// T_i = sum_{j: g[j] <= g[i]} e[j] : 16x32 f16 masked A-tiles times all-ones B.
__global__ __launch_bounds__(BLOCK) void gumbel_logprob(
    const float* __restrict__ logits,
    const float* __restrict__ g,      // [N] f32 (prep wrote into d_out+N)
    const _Float16* __restrict__ e,   // [N] f16 (workspace)
    float* __restrict__ lp) {         // [N] f32 output
    extern __shared__ char smem[];
    float*    sg = (float*)smem;                                   // 2*CHUNK f32
    _Float16* se = (_Float16*)(smem + 2 * CHUNK * sizeof(float));  // 2*CHUNK f16

    const int tid  = threadIdx.x;
    const int lane = tid & 31;
    const int wave = tid >> 5;
    const int m    = lane & 15;
    const int h8   = (lane >> 4) << 3;  // 0 or 8 (K-offset / row-offset)
    const int rowBase = (blockIdx.x * WAVES + wave) * ROWS_PER_WAVE;
    const float gi = g[rowBase + m];

    v16h ones;
#pragma unroll
    for (int t = 0; t < 16; ++t) ones[t] = (_Float16)1.0f;

    v8f acc0 = {};
    v8f acc1 = {};

    // issue one chunk's async copies: 4x b128 of g + 2x b128 of e per thread
    auto issue = [&](int c) {
        const int buf = c & 1;
        const char* gsrc = (const char*)(g + c * CHUNK);
        const char* esrc = (const char*)(e + c * CHUNK);
        char* gdst = (char*)(sg + buf * CHUNK);
        char* edst = (char*)(se + buf * CHUNK);
#if HAVE_ASYNC
#pragma unroll
        for (int k = 0; k < 4; ++k) {
            int off = (tid + k * BLOCK) * 16;
            __builtin_amdgcn_global_load_async_to_lds_b128(
                (__attribute__((address_space(1))) vec_b128*)(gsrc + off),
                (__attribute__((address_space(3))) vec_b128*)(gdst + off), 0, 0);
        }
#pragma unroll
        for (int k = 0; k < 2; ++k) {
            int off = (tid + k * BLOCK) * 16;
            __builtin_amdgcn_global_load_async_to_lds_b128(
                (__attribute__((address_space(1))) vec_b128*)(esrc + off),
                (__attribute__((address_space(3))) vec_b128*)(edst + off), 0, 0);
        }
#else
#pragma unroll
        for (int k = 0; k < 4; ++k) {
            int off = (tid + k * BLOCK) * 16;
            *(float4*)(gdst + off) = *(const float4*)(gsrc + off);
        }
#pragma unroll
        for (int k = 0; k < 2; ++k) {
            int off = (tid + k * BLOCK) * 16;
            *(float4*)(edst + off) = *(const float4*)(esrc + off);
        }
#endif
    };

    issue(0);
    for (int c = 0; c < NCHUNK; ++c) {
        if (c + 1 < NCHUNK) {
            issue(c + 1);
#if HAVE_ASYNC
            __builtin_amdgcn_s_wait_asynccnt(6);  // drain chunk c, keep c+1 inflight
#endif
        } else {
#if HAVE_ASYNC
            __builtin_amdgcn_s_wait_asynccnt(0);
#endif
        }
        __syncthreads();

        const float*    sgc = sg + (c & 1) * CHUNK;
        const _Float16* sec = se + (c & 1) * CHUNK;

        // one 16x32 masked A-tile -> WMMA row-sum into accR.
        // Per lane the tile touches two contiguous 8-element ranges [k1,k1+8)
        // and [k2,k2+8) in both g and e -> 4x ds_load_b128 (g) + 2x (e).
        // Selects are register-only -> v_cndmask_b16, never branches.
        auto tile = [&](int jb, v8f& accR) {
            const int k1 = jb + h8;        // A elements 0..7  (K = k1..k1+7)
            const int k2 = jb + 16 + h8;   // A elements 8..15 (K = k2..k2+7)
            v4f g0 = *(const v4f*)(sgc + k1);
            v4f g1 = *(const v4f*)(sgc + k1 + 4);
            v4f g2 = *(const v4f*)(sgc + k2);
            v4f g3 = *(const v4f*)(sgc + k2 + 4);
            v8h e0 = *(const v8h*)(sec + k1);
            v8h e1 = *(const v8h*)(sec + k2);
            v16h a;
#pragma unroll
            for (int t = 0; t < 4; ++t) {
                a[t]      = (g0[t] <= gi) ? e0[t]     : (_Float16)0.0f;
                a[t + 4]  = (g1[t] <= gi) ? e0[t + 4] : (_Float16)0.0f;
                a[t + 8]  = (g2[t] <= gi) ? e1[t]     : (_Float16)0.0f;
                a[t + 12] = (g3[t] <= gi) ? e1[t + 4] : (_Float16)0.0f;
            }
            accR = __builtin_amdgcn_wmma_f32_16x16x32_f16(
                false, a, false, ones, (short)0, accR, false, false);
        };

#pragma unroll 2
        for (int jb = 0; jb < CHUNK; jb += 64) {
            tile(jb,      acc0);   // independent accumulators break the
            tile(jb + 32, acc1);   // WMMA->WMMA D->C RAW hazard chain
        }
        __syncthreads();
    }

    // D layout: lanes 0-15 VGPR v = row v; lanes 16-31 VGPR v = row v+8,
    // replicated across all N columns -> lanes 0 and 16 each write 8 rows.
    if (m == 0) {
#pragma unroll
        for (int v = 0; v < 8; ++v) {
            int r = rowBase + h8 + v;
            float li = logits[r];
            float T = acc0[v] + acc1[v];
            lp[r] = -logf(expf(-li) * T + 1e-10f);
        }
    }
}

extern "C" void kernel_launch(void* const* d_in, const int* in_sizes, int n_in,
                              void* d_out, int out_size, void* d_ws, size_t ws_size,
                              hipStream_t stream) {
    (void)in_sizes; (void)n_in; (void)out_size; (void)ws_size;
    const float* logits = (const float*)d_in[0];
    const float* noise  = (const float*)d_in[1];
    float* lp   = (float*)d_out;             // output 0: log_prob [N]
    float* gout = (float*)d_out + N_TOTAL;   // output 1: g        [N]
    _Float16* e = (_Float16*)d_ws;           // scratch: exp(logits) f16, 32KB

    gumbel_prep<<<N_TOTAL / 256, 256, 0, stream>>>(logits, noise, gout, e);

    size_t shmem = (size_t)2 * CHUNK * (sizeof(float) + sizeof(_Float16)); // 24 KB
    gumbel_logprob<<<GRID, BLOCK, shmem, stream>>>(logits, gout, e, lp);
}
